// KnowledgeEnhancedGAT_69045894251080
// MI455X (gfx1250) — compile-verified, matrix operand order checked by reading, not run on hardware
//
#include <hip/hip_runtime.h>

// ---------------------------------------------------------------------------
// KnowledgeEnhancedGAT forward for MI455X (gfx1250, wave32, WMMA).
// All GEMMs run through v_wmma_f32_16x16x32_bf16 (bf16 in, f32 accumulate).
// GEMM staging uses GLOBAL_LOAD_ASYNC_TO_LDS_B128 (ASYNCcnt) with LDS
// double-buffering when the builtin is available.
// Problem sizes: B=8, N=1024, D=768, H=4 -> every GEMM dim divides the tile.
// ---------------------------------------------------------------------------

typedef __attribute__((ext_vector_type(16))) __bf16 v16bf;
typedef __attribute__((ext_vector_type(8)))  __bf16 v8bf;
typedef __attribute__((ext_vector_type(8)))  float  v8f;
typedef __attribute__((ext_vector_type(4)))  int    v4i;

#define KB   8
#define KN   1024
#define KD   768
#define KH   4
#define KD4  192
#define KD2  384
#define K2D  1536
#define KHD  3072
#define KROWS (KB * KN)           // 8192
#define ALPHA_LRELU 0.2f
#define K_WEIGHT 0.1f
#define LN_EPS 1e-5f
#define NEG_INF (-3.0e38f)

#if defined(__HIP_DEVICE_COMPILE__) &&                                        \
    __has_builtin(__builtin_amdgcn_global_load_async_to_lds_b128) &&          \
    __has_builtin(__builtin_amdgcn_s_wait_asynccnt)
#define HAS_ASYNC_LDS 1
#else
#define HAS_ASYNC_LDS 0
#endif

// 16-byte global -> LDS transfer (async when supported).
__device__ __forceinline__ void cp16(const __bf16* g, __bf16* l) {
#if HAS_ASYNC_LDS
    __builtin_amdgcn_global_load_async_to_lds_b128(
        (__attribute__((address_space(1))) v4i*)(__bf16*)g,
        (__attribute__((address_space(3))) v4i*)l, 0, 0);
#else
    *(uint4*)l = *(const uint4*)g;
#endif
}

__device__ __forceinline__ void staging_fence() {
#if HAS_ASYNC_LDS
    __builtin_amdgcn_s_wait_asynccnt(0);
#endif
    __syncthreads();
}

// ------------------------------ reductions ---------------------------------
__device__ __forceinline__ float blockSum(float v, float* red) {
#pragma unroll
    for (int o = 16; o > 0; o >>= 1) v += __shfl_down(v, o, 32);
    int lane = threadIdx.x & 31, w = threadIdx.x >> 5;
    if (lane == 0) red[w] = v;
    __syncthreads();
    float r = 0.f;
    if (w == 0) {
        r = (lane < 8) ? red[lane] : 0.f;
#pragma unroll
        for (int o = 4; o > 0; o >>= 1) r += __shfl_down(r, o, 32);
        if (lane == 0) red[0] = r;
    }
    __syncthreads();
    r = red[0];
    __syncthreads();
    return r;
}

__device__ __forceinline__ float blockMax(float v, float* red) {
#pragma unroll
    for (int o = 16; o > 0; o >>= 1) v = fmaxf(v, __shfl_down(v, o, 32));
    int lane = threadIdx.x & 31, w = threadIdx.x >> 5;
    if (lane == 0) red[w] = v;
    __syncthreads();
    float r = NEG_INF;
    if (w == 0) {
        r = (lane < 8) ? red[lane] : NEG_INF;
#pragma unroll
        for (int o = 4; o > 0; o >>= 1) r = fmaxf(r, __shfl_down(r, o, 32));
        if (lane == 0) red[0] = r;
    }
    __syncthreads();
    r = red[0];
    __syncthreads();
    return r;
}

// --------------------------- WMMA fragment load ----------------------------
// 16-bit A-matrix 16x32 layout: lane<16 -> M=lane, K={0..7,16..23};
//                               lane>=16 -> M=lane-16, K={8..15,24..31}.
__device__ __forceinline__ v16bf load_frag(const __bf16* rowBase, int koff) {
    v8bf lo = *(const v8bf*)(rowBase + koff);
    v8bf hi = *(const v8bf*)(rowBase + koff + 16);
    v16bf r;
#pragma unroll
    for (int i = 0; i < 8; ++i) { r[i] = lo[i]; r[8 + i] = hi[i]; }
    return r;
}

// ------------------------------ batched GEMM -------------------------------
// C[z] = A[z] (MxK, row-major, lda) * Bt[z]^T  (Bt stored NxK row-major, ldb)
//        + bias.  f32 accumulate via v_wmma_f32_16x16x32_bf16.
// offA = (z/dA)*sA ; offB = (z%mB)*sB ; offC = z*sC
// Requires: M%128==0, Nn%64==0, K%32==0, lda/ldb multiples of 16.
#define BM 128
#define BNT 64
#define BK 32
__global__ __launch_bounds__(256) void wmma_gemm_bf16(
    const __bf16* __restrict__ A, const __bf16* __restrict__ Bt,
    const float* __restrict__ bias, float* __restrict__ C,
    int M, int Nn, int K, int lda, int ldb, int ldc,
    long sA, int dA, long sB, int mB, long sC) {
    __shared__ __bf16 As[2][BM][BK];
    __shared__ __bf16 Bs[2][BNT][BK];

    const int z = blockIdx.z;
    A  += (long)(z / dA) * sA;
    Bt += (long)(z % mB) * sB;
    C  += (long)z * sC;

    const int tid  = threadIdx.x;
    const int lane = tid & 31;
    const int wave = tid >> 5;
    const int wm = wave >> 1;       // 0..3
    const int wn = wave & 1;        // 0..1
    const int mTile = blockIdx.y * BM;
    const int nTile = blockIdx.x * BNT;

    const int l16  = lane & 15;
    const int koff = (lane < 16) ? 0 : 8;

    v8f acc[2][2];
#pragma unroll
    for (int im = 0; im < 2; ++im)
#pragma unroll
        for (int in = 0; in < 2; ++in)
#pragma unroll
            for (int e = 0; e < 8; ++e) acc[im][in][e] = 0.f;

    // Per-thread staging assignments (16B granules)
    const int ar = tid >> 1;                 // 0..127
    const int ac = (tid & 1) * 16;           // 0 or 16
    const int br = tid >> 2;                 // 0..63
    const int bc = (tid & 3) * 8;            // 0,8,16,24
    const __bf16* aSrc0 = A + (size_t)(mTile + ar) * lda + ac;
    const __bf16* bSrc0 = Bt + (size_t)(nTile + br) * ldb + bc;

    const int nk = K / BK;
    // ---- preload tile 0 into buffer 0 ----
    cp16(aSrc0, &As[0][ar][ac]);
    cp16(aSrc0 + 8, &As[0][ar][ac + 8]);
    cp16(bSrc0, &Bs[0][br][bc]);
    staging_fence();

    int cur = 0;
    for (int kt = 0; kt < nk; ++kt) {
        // ---- stage next tile into the other buffer (overlaps WMMA) ----
        if (kt + 1 < nk) {
            const int k1 = (kt + 1) * BK;
            cp16(aSrc0 + k1, &As[cur ^ 1][ar][ac]);
            cp16(aSrc0 + k1 + 8, &As[cur ^ 1][ar][ac + 8]);
            cp16(bSrc0 + k1, &Bs[cur ^ 1][br][bc]);
        }

        // ---- fragments + WMMA on current buffer ----
        v16bf af[2], bf[2];
#pragma unroll
        for (int im = 0; im < 2; ++im)
            af[im] = load_frag(&As[cur][wm * 32 + im * 16 + l16][0], koff);
#pragma unroll
        for (int in = 0; in < 2; ++in)
            bf[in] = load_frag(&Bs[cur][wn * 32 + in * 16 + l16][0], koff);

#pragma unroll
        for (int im = 0; im < 2; ++im)
#pragma unroll
            for (int in = 0; in < 2; ++in)
                acc[im][in] = __builtin_amdgcn_wmma_f32_16x16x32_bf16(
                    false, af[im], false, bf[in], (short)0, acc[im][in],
                    false, false);

        staging_fence();   // my async loads done + everyone past reads
        cur ^= 1;
    }

    // ---- epilogue: C/D layout -> lanes 0-15: M=v, N=lane; 16-31: M=8+v ----
#pragma unroll
    for (int im = 0; im < 2; ++im) {
        const int mBase = mTile + wm * 32 + im * 16 + ((lane < 16) ? 0 : 8);
#pragma unroll
        for (int in = 0; in < 2; ++in) {
            const int n = nTile + wn * 32 + in * 16 + l16;
            const float bv = bias ? bias[n] : 0.f;
#pragma unroll
            for (int v = 0; v < 8; ++v)
                C[(size_t)(mBase + v) * ldc + n] = acc[im][in][v] + bv;
        }
    }
}

// ------------------------- convert / transpose -----------------------------
__global__ void f32_to_bf16_strided(const float* __restrict__ src,
                                    __bf16* __restrict__ dst,
                                    int rows, int cols, int srcLd, int dstLd,
                                    int dstOff) {
    long total = (long)rows * cols;
    for (long i = (long)blockIdx.x * blockDim.x + threadIdx.x; i < total;
         i += (long)gridDim.x * blockDim.x) {
        long r = i / cols, c = i % cols;
        dst[r * dstLd + dstOff + c] = (__bf16)src[r * srcLd + c];
    }
}

__global__ void tconv_kernel(const float* __restrict__ src,
                             __bf16* __restrict__ dst, int rows, int cols,
                             long sS, long sD) {
    __shared__ float t[32][33];
    src += (long)blockIdx.z * sS;
    dst += (long)blockIdx.z * sD;
    int r0 = blockIdx.y * 32, c0 = blockIdx.x * 32;
    for (int j = threadIdx.y; j < 32; j += 8) {
        int r = r0 + j, c = c0 + threadIdx.x;
        t[j][threadIdx.x] =
            (r < rows && c < cols) ? src[(size_t)r * cols + c] : 0.f;
    }
    __syncthreads();
    for (int j = threadIdx.y; j < 32; j += 8) {
        int c = c0 + j, r = r0 + threadIdx.x;
        if (c < cols && r < rows)
            dst[(size_t)c * rows + r] = (__bf16)t[threadIdx.x][j];
    }
}

// ---------------------- gate kernels (LN + MLP head) -----------------------
// mode 0: confidence gate  (gate = sigmoid((logit+gate_bias)/max(temp,.1)))
//         gate_out = gate, aux_out = gate * mul[row]   (gated polarity)
// mode 1: dynamic gate     (g = sigmoid(logit) * mul[row])
//         gate_out = g (d_out), aux_out = g (workspace)
__global__ __launch_bounds__(256) void gate_kernel(
    const float* __restrict__ x, int cols, const float* __restrict__ lng,
    const float* __restrict__ lnb, const float* __restrict__ w2,
    const float* __restrict__ b2, const float* __restrict__ gate_bias,
    const float* __restrict__ temp, const float* __restrict__ mul,
    float* __restrict__ gate_out, float* __restrict__ aux_out, int mode) {
    __shared__ float red[8];
    const int row = blockIdx.x;
    const float* xr = x + (size_t)row * cols;

    float v0 = 0.f, v1 = 0.f;
    int c0 = threadIdx.x, c1 = threadIdx.x + 256;
    if (c0 < cols) v0 = xr[c0];
    if (c1 < cols) v1 = xr[c1];
    float cnt = (float)cols;
    float mean = blockSum(((c0 < cols) ? v0 : 0.f) + ((c1 < cols) ? v1 : 0.f),
                          red) / cnt;
    float d0 = (c0 < cols) ? (v0 - mean) : 0.f;
    float d1 = (c1 < cols) ? (v1 - mean) : 0.f;
    float var = blockSum(d0 * d0 + d1 * d1, red) / cnt;
    float inv = rsqrtf(var + LN_EPS);

    float dot = 0.f;
    if (c0 < cols) {
        float y = fmaxf(d0 * inv * lng[c0] + lnb[c0], 0.f);
        dot += y * w2[c0];
    }
    if (c1 < cols) {
        float y = fmaxf(d1 * inv * lng[c1] + lnb[c1], 0.f);
        dot += y * w2[c1];
    }
    float logit = blockSum(dot, red) + b2[0];

    if (threadIdx.x == 0) {
        if (mode == 0) {
            float t = fmaxf(temp[0], 0.1f);
            float gate = 1.f / (1.f + __expf(-((logit + gate_bias[0]) / t)));
            gate_out[row] = gate;
            aux_out[row] = gate * mul[row];
        } else {
            float g = (1.f / (1.f + __expf(-logit))) * mul[row];
            gate_out[row] = g;
            aux_out[row] = g;
        }
    }
}

// --------------------- small element-wise helper kernels -------------------
__global__ void pe_h1_kernel(const float* __restrict__ pol,
                             const float* __restrict__ w1,
                             const float* __restrict__ b1,
                             __bf16* __restrict__ out) {
    long total = (long)KROWS * KD4;
    for (long i = (long)blockIdx.x * blockDim.x + threadIdx.x; i < total;
         i += (long)gridDim.x * blockDim.x) {
        int c = (int)(i % KD4);
        long r = i / KD4;
        float v = pol[r] * w1[c] + b1[c];
        out[i] = (__bf16)fmaxf(v, 0.f);
    }
}

__global__ void fuse_kernel(const float* __restrict__ h,
                            const float* __restrict__ pe,
                            const float* __restrict__ g,
                            __bf16* __restrict__ out) {
    long total = (long)KROWS * KD;
    for (long i = (long)blockIdx.x * blockDim.x + threadIdx.x; i < total;
         i += (long)gridDim.x * blockDim.x) {
        long r = i / KD;
        float gv = g[r];
        out[i] = (__bf16)((1.f - gv) * h[i] + gv * pe[i]);
    }
}

// LN with residual.  mode 0: enh = LN(x) + res.  mode 1: out = LN(x + res).
__global__ __launch_bounds__(256) void ln_res_kernel(
    const float* __restrict__ x, const float* __restrict__ res,
    const float* __restrict__ lng, const float* __restrict__ lnb,
    float* __restrict__ outf, __bf16* __restrict__ outb, int mode) {
    __shared__ float red[8];
    const int row = blockIdx.x;
    const float* xr = x + (size_t)row * KD;
    const float* rr = res + (size_t)row * KD;
    float v[3];
#pragma unroll
    for (int t = 0; t < 3; ++t) {
        int c = threadIdx.x + t * 256;
        v[t] = xr[c] + ((mode == 1) ? rr[c] : 0.f);
    }
    float mean = blockSum(v[0] + v[1] + v[2], red) * (1.f / KD);
    float sq = 0.f;
#pragma unroll
    for (int t = 0; t < 3; ++t) { v[t] -= mean; sq += v[t] * v[t]; }
    float inv = rsqrtf(blockSum(sq, red) * (1.f / KD) + LN_EPS);
#pragma unroll
    for (int t = 0; t < 3; ++t) {
        int c = threadIdx.x + t * 256;
        float y = v[t] * inv * lng[c] + lnb[c];
        if (mode == 0) y += rr[c];
        outf[(size_t)row * KD + c] = y;
        if (outb) outb[(size_t)row * KD + c] = (__bf16)y;
    }
}

// e_src / e_dst: per-row dot of Wh with a_src / a_dst (one wave per row).
__global__ __launch_bounds__(256) void dots_kernel(
    const float* __restrict__ Wh, const float* __restrict__ a_src,
    const float* __restrict__ a_dst, float* __restrict__ es,
    float* __restrict__ ed) {
    const long row = (long)blockIdx.x * 8 + (threadIdx.x >> 5);
    const int lane = threadIdx.x & 31;
    if (row >= (long)KB * KH * KN) return;
    const int h = (int)((row / KN) % KH);
    const float* wr = Wh + row * KD;
    const float* as = a_src + (size_t)h * KD;
    const float* ad = a_dst + (size_t)h * KD;
    float s = 0.f, d = 0.f;
    for (int c = lane; c < KD; c += 32) {
        float w = wr[c];
        s += w * as[c];
        d += w * ad[c];
    }
#pragma unroll
    for (int o = 16; o > 0; o >>= 1) {
        s += __shfl_down(s, o, 32);
        d += __shfl_down(d, o, 32);
    }
    if (lane == 0) { es[row] = s; ed[row] = d; }
}

// Attention scores + softmax over j.  Writes attn f32 (d_out) + bf16 (ws).
__global__ __launch_bounds__(256) void attn_kernel(
    const float* __restrict__ es, const float* __restrict__ ed,
    const float* __restrict__ gp, const float* __restrict__ adj,
    const float* __restrict__ amask, float* __restrict__ attn_f,
    __bf16* __restrict__ attn_b) {
    __shared__ float red[8];
    const int i = blockIdx.x, h = blockIdx.y, b = blockIdx.z;
    const long rowIdx = ((long)(b * KH + h) * KN + i);
    const float esv = es[rowIdx];
    const float gpi = gp[(size_t)b * KN + i] * K_WEIGHT;
    const float* adjR = adj + ((size_t)b * KN + i) * KN;
    const float* edR = ed + (long)(b * KH + h) * KN;
    const float* gpR = gp + (size_t)b * KN;
    const float* amR = amask + (size_t)b * KN;

    float vals[4];
    float mx = NEG_INF;
#pragma unroll
    for (int t = 0; t < 4; ++t) {
        int j = t * 256 + threadIdx.x;
        float e = esv + edR[j];
        e = (e > 0.f) ? e : ALPHA_LRELU * e;
        e += gpi * gpR[j];
        bool keep = (adjR[j] != 0.f) && (amR[j] != 0.f);
        e = keep ? e : NEG_INF;
        vals[t] = e;
        mx = fmaxf(mx, e);
    }
    mx = blockMax(mx, red);
    float s = 0.f;
#pragma unroll
    for (int t = 0; t < 4; ++t) {
        float p = (vals[t] <= NEG_INF) ? 0.f : __expf(vals[t] - mx);
        vals[t] = p;
        s += p;
    }
    s = blockSum(s, red);
    float inv = (s > 0.f) ? (1.f / s) : 0.f;
    float* of = attn_f + rowIdx * KN;
    __bf16* ob = attn_b + rowIdx * KN;
#pragma unroll
    for (int t = 0; t < 4; ++t) {
        int j = t * 256 + threadIdx.x;
        float a = vals[t] * inv;
        of[j] = a;
        ob[j] = (__bf16)a;
    }
}

// elu(hp) and scatter [B,H,N,D] -> [B,N,H*D] bf16.
__global__ void elu_cat_kernel(const float* __restrict__ hp,
                               __bf16* __restrict__ cat) {
    long total = (long)KB * KH * KN * KD;
    for (long i = (long)blockIdx.x * blockDim.x + threadIdx.x; i < total;
         i += (long)gridDim.x * blockDim.x) {
        int d = (int)(i % KD);
        long t = i / KD;
        int n = (int)(t % KN);
        t /= KN;
        int h = (int)(t % KH);
        int b = (int)(t / KH);
        float x = hp[i];
        float e = (x > 0.f) ? x : (__expf(x) - 1.f);
        cat[((size_t)b * KN + n) * KHD + (size_t)h * KD + d] = (__bf16)e;
    }
}

// ------------------------------- host side ---------------------------------
static inline size_t alignUp(size_t v) { return (v + 255) & ~(size_t)255; }

extern "C" void kernel_launch(void* const* d_in, const int* in_sizes, int n_in,
                              void* d_out, int out_size, void* d_ws,
                              size_t ws_size, hipStream_t stream) {
    (void)in_sizes; (void)n_in; (void)out_size; (void)ws_size;
    // ---- inputs (setup_inputs dict order, params flattened in dict order)
    const float* hs    = (const float*)d_in[0];
    const float* pol   = (const float*)d_in[1];
    const float* adj   = (const float*)d_in[2];
    const float* amask = (const float*)d_in[3];
    const float* cov   = (const float*)d_in[4];
    // conf_params
    const float* cW1 = (const float*)d_in[5];
    const float* cb1 = (const float*)d_in[6];
    const float* clg = (const float*)d_in[7];
    const float* clb = (const float*)d_in[8];
    const float* cW2 = (const float*)d_in[9];
    const float* cb2 = (const float*)d_in[10];
    const float* cgb = (const float*)d_in[11];
    const float* ctemp = (const float*)d_in[12];
    // dyn_params
    const float* peW1 = (const float*)d_in[13];
    const float* peb1 = (const float*)d_in[14];
    const float* peW2 = (const float*)d_in[15];
    const float* peb2 = (const float*)d_in[16];
    const float* gW1  = (const float*)d_in[17];
    const float* gb1  = (const float*)d_in[18];
    const float* glg  = (const float*)d_in[19];
    const float* glb  = (const float*)d_in[20];
    const float* gW2  = (const float*)d_in[21];
    const float* gb2  = (const float*)d_in[22];
    const float* opW  = (const float*)d_in[23];
    const float* opb  = (const float*)d_in[24];
    const float* oplg = (const float*)d_in[25];
    const float* oplb = (const float*)d_in[26];
    // gat_params
    const float* gatW  = (const float*)d_in[27];
    const float* aSrc  = (const float*)d_in[28];
    const float* aDst  = (const float*)d_in[29];
    const float* projW = (const float*)d_in[30];
    const float* projb = (const float*)d_in[31];
    const float* flg   = (const float*)d_in[32];
    const float* flb   = (const float*)d_in[33];

    // ---- outputs (tuple: out, attn, conf_gate, g) concatenated f32
    float* outMain = (float*)d_out;                              // [B,N,D]
    float* outAttn = outMain + (size_t)KB * KN * KD;             // [B,H,N,N]
    float* outConf = outAttn + (size_t)KB * KH * KN * KN;        // [B,N]
    float* outG    = outConf + (size_t)KB * KN;                  // [B,N]

    // ---- workspace carve (assumes ws_size >= ~380 MB)
    char* ws = (char*)d_ws;
    size_t cur = 0;
    auto alloc = [&](size_t bytes) -> char* {
        char* p = ws + cur;
        cur = alignUp(cur + bytes);
        return p;
    };
    __bf16* gi      = (__bf16*)alloc((size_t)KROWS * K2D * 2);   // [hs|pe] bf16
    __bf16* W1t     = (__bf16*)alloc((size_t)KD4 * KD * 2);
    float*  tmp1    = (float*)alloc((size_t)KROWS * KD4 * 4);
    float*  gpol    = (float*)alloc((size_t)KROWS * 4);
    __bf16* h1pe    = (__bf16*)alloc((size_t)KROWS * KD4 * 2);
    __bf16* peW2t   = (__bf16*)alloc((size_t)KD * KD4 * 2);
    float*  pe      = (float*)alloc((size_t)KROWS * KD * 4);     // aliased later
    __bf16* gW1t    = (__bf16*)alloc((size_t)KD2 * K2D * 2);
    float*  tmp2    = (float*)alloc((size_t)KROWS * KD2 * 4);
    float*  gws     = (float*)alloc((size_t)KROWS * 4);
    __bf16* fusedB  = (__bf16*)alloc((size_t)KROWS * KD * 2);
    __bf16* opWt    = (__bf16*)alloc((size_t)KD * KD * 2);
    float*  tmp3    = (float*)alloc((size_t)KROWS * KD * 4);
    float*  enh     = (float*)alloc((size_t)KROWS * KD * 4);
    __bf16* enhB    = (__bf16*)alloc((size_t)KROWS * KD * 2);
    __bf16* Wt      = (__bf16*)alloc((size_t)KH * KD * KD * 2);
    float*  Wh      = (float*)alloc((size_t)KB * KH * KN * KD * 4); // alias hp
    float*  eSrc    = (float*)alloc((size_t)KB * KH * KN * 4);
    float*  eDst    = (float*)alloc((size_t)KB * KH * KN * 4);
    __bf16* WhT     = (__bf16*)alloc((size_t)KB * KH * KD * KN * 2); // alias cat
    __bf16* attnB   = (__bf16*)alloc((size_t)KB * KH * KN * KN * 2);
    __bf16* projWt  = (__bf16*)alloc((size_t)KD * KHD * 2);
    float*  hp      = Wh;            // reuse after WhT + dots are done
    __bf16* catB    = WhT;           // reuse after hp GEMM is done
    float*  outTmp  = pe;            // reuse after fusion is done

    const dim3 blk256(256);
    const dim3 blkT(32, 8);
    const long sWh = (long)KN * KD;       // per (b,h) Wh stride
    const long sWd = (long)KD * KD;       // per-head weight stride

    // 1) hidden -> gi[:, :768] bf16
    f32_to_bf16_strided<<<4096, blk256, 0, stream>>>(hs, gi, KROWS, KD, KD,
                                                     K2D, 0);
    // 2) confidence: W1^T, GEMM, LN+MLP gate
    tconv_kernel<<<dim3(KD4 / 32, KD / 32, 1), blkT, 0, stream>>>(
        cW1, W1t, KD, KD4, 0, 0);
    wmma_gemm_bf16<<<dim3(KD4 / 64, KROWS / 128, 1), blk256, 0, stream>>>(
        gi, W1t, cb1, tmp1, KROWS, KD4, KD, K2D, KD, KD4, 0, 1, 0, 1, 0);
    gate_kernel<<<KROWS, blk256, 0, stream>>>(tmp1, KD4, clg, clb, cW2, cb2,
                                              cgb, ctemp, pol, outConf, gpol,
                                              0);
    // 3) polarity embedding: h1 = relu(pol*W1+b1); pe = h1 @ peW2 + b2
    pe_h1_kernel<<<2048, blk256, 0, stream>>>(pol, peW1, peb1, h1pe);
    tconv_kernel<<<dim3(KD / 32, KD4 / 32, 1), blkT, 0, stream>>>(
        peW2, peW2t, KD4, KD, 0, 0);
    wmma_gemm_bf16<<<dim3(KD / 64, KROWS / 128, 1), blk256, 0, stream>>>(
        h1pe, peW2t, peb2, pe, KROWS, KD, KD4, KD4, KD4, KD, 0, 1, 0, 1, 0);
    // 4) gi[:, 768:] = bf16(pe); dynamic gate g
    f32_to_bf16_strided<<<4096, blk256, 0, stream>>>(pe, gi, KROWS, KD, KD,
                                                     K2D, KD);
    tconv_kernel<<<dim3(KD2 / 32, K2D / 32, 1), blkT, 0, stream>>>(
        gW1, gW1t, K2D, KD2, 0, 0);
    wmma_gemm_bf16<<<dim3(KD2 / 64, KROWS / 128, 1), blk256, 0, stream>>>(
        gi, gW1t, gb1, tmp2, KROWS, KD2, K2D, K2D, K2D, KD2, 0, 1, 0, 1, 0);
    gate_kernel<<<KROWS, blk256, 0, stream>>>(tmp2, KD2, glg, glb, gW2, gb2,
                                              nullptr, nullptr, cov, outG,
                                              gws, 1);
    // 5) fused = (1-g)h + g*pe ; enh = LN(fused@opW + b) + h
    fuse_kernel<<<4096, blk256, 0, stream>>>(hs, pe, gws, fusedB);
    tconv_kernel<<<dim3(KD / 32, KD / 32, 1), blkT, 0, stream>>>(opW, opWt,
                                                                 KD, KD, 0, 0);
    wmma_gemm_bf16<<<dim3(KD / 64, KROWS / 128, 1), blk256, 0, stream>>>(
        fusedB, opWt, opb, tmp3, KROWS, KD, KD, KD, KD, KD, 0, 1, 0, 1, 0);
    ln_res_kernel<<<KROWS, blk256, 0, stream>>>(tmp3, hs, oplg, oplb, enh,
                                                enhB, 0);
    // 6) GAT: Wh[b,h] = enh[b] @ W[h]
    tconv_kernel<<<dim3(KD / 32, KD / 32, KH), blkT, 0, stream>>>(
        gatW, Wt, KD, KD, sWd, sWd);
    wmma_gemm_bf16<<<dim3(KD / 64, KN / 128, KB * KH), blk256, 0, stream>>>(
        enhB, Wt, nullptr, Wh, KN, KD, KD, KD, KD, KD,
        sWh, KH, sWd, KH, sWh);
    // 7) attention logits + softmax
    dots_kernel<<<(KB * KH * KN) / 8, blk256, 0, stream>>>(Wh, aSrc, aDst,
                                                           eSrc, eDst);
    tconv_kernel<<<dim3(KD / 32, KN / 32, KB * KH), blkT, 0, stream>>>(
        Wh, WhT, KN, KD, sWh, sWh);
    attn_kernel<<<dim3(KN, KH, KB), blk256, 0, stream>>>(
        eSrc, eDst, gpol, adj, amask, outAttn, attnB);
    // 8) hp = attn @ Wh  (Wh reused as hp output)
    wmma_gemm_bf16<<<dim3(KD / 64, KN / 128, KB * KH), blk256, 0, stream>>>(
        attnB, WhT, nullptr, hp, KN, KD, KN, KN, KN, KD,
        (long)KN * KN, 1, (long)KD * KN, 1 << 30, sWh);
    // 9) elu + concat heads; out = LN(cat@projW + b + enh)
    elu_cat_kernel<<<8192, blk256, 0, stream>>>(hp, catB);
    tconv_kernel<<<dim3(KD / 32, KHD / 32, 1), blkT, 0, stream>>>(
        projW, projWt, KHD, KD, 0, 0);
    wmma_gemm_bf16<<<dim3(KD / 64, KROWS / 128, 1), blk256, 0, stream>>>(
        catB, projWt, projb, outTmp, KROWS, KD, KHD, KHD, KHD, KD,
        0, 1, 0, 1, 0);
    ln_res_kernel<<<KROWS, blk256, 0, stream>>>(outTmp, enh, flg, flb,
                                                outMain, nullptr, 1);
}
